// SingleScaleDeformAttn_21672404976132
// MI455X (gfx1250) — matrix operand order repaired and checked by reading, not compiled, not run on hardware
//
#include <hip/hip_runtime.h>
#include <hip/hip_bf16.h>

// ---------------------------------------------------------------------------
// Multi-scale deformable attention for MI455X (gfx1250, wave32, WMMA).
//
// Pipeline:
//   1) pack_w16:      W_value / W_out (256x256 f32) -> f16, pre-swizzled into
//                     the wave32 WMMA B-operand layout (contiguous 32B/lane).
//   2) gemm256_wmma<HALF_OUT=1>: value = input_flatten @ W_value + b_value,
//                     f32 accum via v_wmma_f32_16x16x32_f16, stored f16 so the
//                     107.5MB value tensor fits in the 192MB L2 for gathers.
//   3) ms_deform_sample: fused offset/attn projection (+softmax) and bilinear
//                     gather-accumulate -> pre_out (43200 x 256, f32).
//   4) gemm256_wmma<HALF_OUT=0>: out = pre_out @ W_out + b_out.
// ---------------------------------------------------------------------------

typedef __attribute__((ext_vector_type(16))) _Float16 v16h;
typedef __attribute__((ext_vector_type(8)))  float    v8f;

#define D_MODEL   256
#define N_HEADS   8
#define HEAD_DIM  32
#define N_LEVELS  3
#define N_POINTS  2
#define LEN_IN    13125
#define BS        16
#define NQ        900
#define NROW_Q    (BS * NQ * N_LEVELS)   // 43200
#define NROW_V    (BS * LEN_IN)          // 210000

// ---------------------------------------------------------------------------
// Pack a 256x256 row-major f32 weight matrix into f16 WMMA-B layout:
//   b[e] of lane L in n-tile t, k-block kb  =  W[kb*32 + (L<16?0:16) + e][t*16 + (L&15)]
// stored so each lane's 16-half fragment is one contiguous 32-byte load:
//   Bp[(((t*8 + kb)*32 + lane)*16) + e]
// ---------------------------------------------------------------------------
__global__ void pack_w16(const float* __restrict__ W, _Float16* __restrict__ Bp) {
    int idx  = blockIdx.x * blockDim.x + threadIdx.x;   // 0 .. 65535
    int e    = idx & 15;
    int lane = (idx >> 4) & 31;
    int kb   = (idx >> 9) & 7;
    int t    = idx >> 12;
    int k    = kb * 32 + ((lane < 16) ? 0 : 16) + e;
    int n    = t * 16 + (lane & 15);
    Bp[idx] = (_Float16)W[(size_t)k * 256 + n];
}

// ---------------------------------------------------------------------------
// C[M x 256] = A[M x 256] @ B[256 x 256] + bias, f32 accum, WMMA f16 inputs.
// Block = 256 threads = 8 waves; block owns one 16-row m-tile and all 256
// columns (wave w -> n-tiles w and w+8). A tile staged once in LDS as f16.
// ---------------------------------------------------------------------------
template <bool HALF_OUT>
__global__ __launch_bounds__(256) void gemm256_wmma(
    const float* __restrict__ A,        // M x 256, f32, row-major
    const _Float16* __restrict__ Bp,    // packed f16 B (pack_w16 layout)
    const float* __restrict__ bias,     // 256
    void* __restrict__ Cout)            // M x 256 (f16 or f32)
{
    constexpr int K = 256, N = 256;
    __shared__ _Float16 As[16][K + 8];      // +8 halves pad, rows stay 16B aligned

    const int tid  = threadIdx.x;
    const int wave = tid >> 5;
    const int lane = tid & 31;
    const int m0   = blockIdx.x * 16;

    // Stage A tile (16 x 256) as f16: thread t -> row t/16, 16 cols.
    {
        int r  = tid >> 4;
        int c0 = (tid & 15) * 16;
        const float* src = A + (size_t)(m0 + r) * K + c0;
        #pragma unroll
        for (int i = 0; i < 16; ++i) As[r][c0 + i] = (_Float16)src[i];
    }
    __syncthreads();

    const int t0    = wave;          // first n-tile
    const int t1    = wave + 8;      // second n-tile
    const int mrow  = lane & 15;
    const int kbase = (lane < 16) ? 0 : 16;  // A-fragment K sub-offset

    v8f acc0 = {};
    v8f acc1 = {};

    #pragma unroll
    for (int kb = 0; kb < 8; ++kb) {
        const int k0 = kb * 32;
        // A fragment: K = k0+kbase+e (contiguous, matches packed B)
        v16h a;
        #pragma unroll
        for (int e = 0; e < 16; ++e)
            a[e] = As[mrow][k0 + kbase + e];

        const v16h b0 = *(const v16h*)(Bp + (((size_t)(t0 * 8 + kb) * 32 + lane) << 4));
        const v16h b1 = *(const v16h*)(Bp + (((size_t)(t1 * 8 + kb) * 32 + lane) << 4));

        acc0 = __builtin_amdgcn_wmma_f32_16x16x32_f16(false, a, false, b0,
                                                      (short)0, acc0, false, false);
        acc1 = __builtin_amdgcn_wmma_f32_16x16x32_f16(false, a, false, b1,
                                                      (short)0, acc1, false, false);
    }

    // D layout: element r -> row m0 + r + (lane<16?0:8), col = tile*16 + (lane&15)
    const int nc0   = t0 * 16 + (lane & 15);
    const int nc1   = t1 * 16 + (lane & 15);
    const int rbase = m0 + ((lane < 16) ? 0 : 8);
    const float bb0 = bias[nc0];
    const float bb1 = bias[nc1];

    #pragma unroll
    for (int r = 0; r < 8; ++r) {
        const size_t row = (size_t)(rbase + r);
        const float v0 = acc0[r] + bb0;
        const float v1 = acc1[r] + bb1;
        if (HALF_OUT) {
            ((_Float16*)Cout)[row * N + nc0] = (_Float16)v0;
            ((_Float16*)Cout)[row * N + nc1] = (_Float16)v1;
        } else {
            ((float*)Cout)[row * N + nc0] = v0;
            ((float*)Cout)[row * N + nc1] = v1;
        }
    }
}

// ---------------------------------------------------------------------------
// Fused: offset/attn projection + softmax + bilinear gather-accumulate.
// One block per (b, q, f) row; 256 threads = 8 head-groups x 32 lanes (d).
// value is f16 (fits in L2), layout (b, pos, head, d) -> coalesced 64B/tap.
// ---------------------------------------------------------------------------
__global__ __launch_bounds__(256) void ms_deform_sample(
    const float* __restrict__ query,     // (43200, 256)
    const float* __restrict__ refpts,    // (43200, 2)
    const float* __restrict__ W_off,     // (256, 96)
    const float* __restrict__ b_off,     // (96)
    const float* __restrict__ W_attn,    // (256, 48)
    const float* __restrict__ b_attn,    // (48)
    const _Float16* __restrict__ value,  // (16, 13125, 8, 32) f16
    float* __restrict__ pre_out)         // (43200, 256)
{
    __shared__ float qs[256];
    __shared__ float offs[96];
    __shared__ float attw[48];

    const int tid = threadIdx.x;
    const int idx = blockIdx.x;                 // (b*900 + q)*3 + f
    const int b   = idx / (NQ * N_LEVELS);

    qs[tid] = query[(size_t)idx * 256 + tid];
    __syncthreads();

    // offsets (96) and attn logits (48): skinny projection, VALU + L2-hot W.
    if (tid < 96) {
        float a = b_off[tid];
        #pragma unroll 8
        for (int k = 0; k < 256; ++k) a = fmaf(qs[k], W_off[k * 96 + tid], a);
        offs[tid] = a;
    } else if (tid < 144) {
        const int j = tid - 96;
        float a = b_attn[j];
        #pragma unroll 8
        for (int k = 0; k < 256; ++k) a = fmaf(qs[k], W_attn[k * 48 + j], a);
        attw[j] = a;
    }
    __syncthreads();

    // softmax over nl*np = 6 per head (8 threads, one per head)
    if (tid < 8) {
        float m = -1e30f;
        #pragma unroll
        for (int j = 0; j < 6; ++j) m = fmaxf(m, attw[tid * 6 + j]);
        float e[6], s = 0.f;
        #pragma unroll
        for (int j = 0; j < 6; ++j) { e[j] = __expf(attw[tid * 6 + j] - m); s += e[j]; }
        const float inv = 1.0f / s;
        #pragma unroll
        for (int j = 0; j < 6; ++j) attw[tid * 6 + j] = e[j] * inv;
    }
    __syncthreads();

    const int h = tid >> 5;
    const int d = tid & 31;
    const float rx = refpts[(size_t)idx * 2 + 0];
    const float ry = refpts[(size_t)idx * 2 + 1];

    const int   HW[3]   = {100, 50, 25};
    const int   BASE[3] = {0, 10000, 12500};

    float acc = 0.f;
    #pragma unroll
    for (int l = 0; l < 3; ++l) {
        const int   Wi = HW[l];
        const float Wl = (float)Wi;
        const float Hl = Wl;                    // square pyramid levels
        const _Float16* vb =
            value + ((((size_t)b * LEN_IN + BASE[l]) * N_HEADS + h) << 5) + d;
        #pragma unroll
        for (int p = 0; p < 2; ++p) {
            const float ox = offs[h * 12 + l * 4 + p * 2 + 0];
            const float oy = offs[h * 12 + l * 4 + p * 2 + 1];
            const float aw = attw[h * 6 + l * 2 + p];
            const float x  = (rx + ox / Wl) * Wl - 0.5f;
            const float y  = (ry + oy / Hl) * Hl - 0.5f;
            const float x0 = floorf(x);
            const float y0 = floorf(y);
            float s = 0.f;
            #pragma unroll
            for (int dy = 0; dy < 2; ++dy) {
                #pragma unroll
                for (int dx = 0; dx < 2; ++dx) {
                    const float xi = x0 + (float)dx;
                    const float yi = y0 + (float)dy;
                    float wgt = (1.f - fabsf(x - xi)) * (1.f - fabsf(y - yi));
                    const bool valid = (xi >= 0.f) && (xi <= Wl - 1.f) &&
                                       (yi >= 0.f) && (yi <= Hl - 1.f);
                    const int xc = (int)fminf(fmaxf(xi, 0.f), Wl - 1.f);
                    const int yc = (int)fminf(fmaxf(yi, 0.f), Hl - 1.f);
                    const float g = (float)vb[(size_t)(yc * Wi + xc) * (N_HEADS * HEAD_DIM)];
                    s = fmaf(valid ? wgt : 0.f, g, s);
                }
            }
            acc = fmaf(aw, s, acc);
        }
    }
    pre_out[(size_t)idx * 256 + tid] = acc;
}

// ---------------------------------------------------------------------------
extern "C" void kernel_launch(void* const* d_in, const int* in_sizes, int n_in,
                              void* d_out, int out_size, void* d_ws, size_t ws_size,
                              hipStream_t stream) {
    (void)in_sizes; (void)n_in; (void)out_size; (void)ws_size;

    const float* query         = (const float*)d_in[0];
    const float* refpts        = (const float*)d_in[1];
    const float* input_flatten = (const float*)d_in[2];
    /* d_in[3] = input_spatial_shapes (int32) — static, hardcoded */
    const float* W_value = (const float*)d_in[4];
    const float* b_value = (const float*)d_in[5];
    const float* W_off   = (const float*)d_in[6];
    const float* b_off   = (const float*)d_in[7];
    const float* W_attn  = (const float*)d_in[8];
    const float* b_attn  = (const float*)d_in[9];
    const float* W_out   = (const float*)d_in[10];
    const float* b_out   = (const float*)d_in[11];
    float* out = (float*)d_out;

    // Workspace layout (all offsets 256B aligned):
    //   value_f16 : 210000*256 halves = 107,520,000 B
    //   Wv packed : 65536 halves      =     131,072 B
    //   Wo packed : 65536 halves      =     131,072 B
    //   pre_out   : 43200*256 f32     =  44,236,800 B    (total ~145 MB)
    char* ws = (char*)d_ws;
    const size_t VAL_BYTES = (size_t)NROW_V * D_MODEL * sizeof(_Float16); // 107520000
    _Float16* value_f16 = (_Float16*)ws;
    _Float16* wv_packed = (_Float16*)(ws + VAL_BYTES);
    _Float16* wo_packed = (_Float16*)(ws + VAL_BYTES + 131072);
    float*    pre_out   = (float*)   (ws + VAL_BYTES + 262144);

    // 1) pack weights into WMMA-B layout (f16)
    pack_w16<<<256, 256, 0, stream>>>(W_value, wv_packed);
    pack_w16<<<256, 256, 0, stream>>>(W_out,   wo_packed);

    // 2) value projection -> f16 value tensor (fits in 192MB L2 for gathers)
    gemm256_wmma<true><<<NROW_V / 16, 256, 0, stream>>>(
        input_flatten, wv_packed, b_value, (void*)value_f16);

    // 3) fused offsets/attn/softmax + bilinear gather
    ms_deform_sample<<<NROW_Q, 256, 0, stream>>>(
        query, refpts, W_off, b_off, W_attn, b_attn, value_f16, pre_out);

    // 4) output projection -> d_out (f32)
    gemm256_wmma<false><<<NROW_Q / 16, 256, 0, stream>>>(
        pre_out, wo_packed, b_out, (void*)out);
}
